// PointPillarsEncoder_41497974014041
// MI455X (gfx1250) — matrix operand order repaired and checked by reading, not compiled, not run on hardware
//
#include <hip/hip_runtime.h>

// ---------------------------------------------------------------------------
// PointPillars encoder for MI455X (gfx1250, wave32, WMMA), NHWC fp16 interior.
//   1) pillarize: scatter points into dense [65536][32][7] + per-cell counter
//   2) pointnet:  7->32 per-lane MLP, 32->64 via v_wmma_f32_16x16x32_f16,
//                 BN+ReLU, max over 32 slots, write BEV NHWC fp16 [256][256][64]
//   3) 4x conv3x3+BN+ReLU implicit GEMM:
//        - full 3x66xCIN NHWC input patch staged in LDS once per block
//        - A frags: two ds_load_b128 (channel-contiguous, bank-padded)
//        - B frags: fragment-order packed weights, coalesced global b128
//        - no barriers inside the K loop
// ---------------------------------------------------------------------------

#define NXg 256
#define NYg 256
#define NCELLSg (NXg * NYg)
#define MAXPTS 32
#define PDIM 7

typedef __attribute__((ext_vector_type(8)))  _Float16 h8;
typedef __attribute__((ext_vector_type(16))) _Float16 h16;
typedef __attribute__((ext_vector_type(8)))  float    v8f;

static __device__ __forceinline__ h16 frag_combine(h8 lo, h8 hv) {
  h16 r;
#pragma unroll
  for (int i = 0; i < 8; ++i) { r[i] = lo[i]; r[i + 8] = hv[i]; }
  return r;
}

// A-matrix fragment (16-bit, 16x32): lane holds row M=lane%16 (hi = lane>>4).
// halves 0..15 = K hi*16 .. hi*16+15
static __device__ __forceinline__ h16 ldA(const _Float16* row, int hi) {
  h8 lo = *(const h8*)(row + hi * 16);
  h8 hv = *(const h8*)(row + hi * 16 + 8);
  return frag_combine(lo, hv);
}

// B-matrix fragment (16-bit, 32x16): lane holds col N=lane%16,
// halves 0..15 = K hi*16 .. hi*16+15 (stored [n][k] contiguous in k).
static __device__ __forceinline__ h16 ldB(const _Float16* row, int hi) {
  h8 lo = *(const h8*)(row + hi * 16);
  h8 hv = *(const h8*)(row + hi * 16 + 8);
  return frag_combine(lo, hv);
}

// ---------------------------------------------------------------------------
// Kernel 1: pillarize
// ---------------------------------------------------------------------------
__global__ __launch_bounds__(256) void pillarize_kernel(
    const float* __restrict__ pts, int n, int* __restrict__ cnt,
    float* __restrict__ dense) {
  int i = blockIdx.x * 256 + threadIdx.x;
  if (i >= n) return;
  float x = pts[i * 4 + 0];
  float y = pts[i * 4 + 1];
  float z = pts[i * 4 + 2];
  float it = pts[i * 4 + 3];
  if (!(x >= -51.2f && x < 51.2f && y >= -51.2f && y < 51.2f)) return;
  int xi = (int)floorf((x + 51.2f) * 2.5f);
  int yi = (int)floorf((y + 51.2f) * 2.5f);
  xi = min(max(xi, 0), NXg - 1);
  yi = min(max(yi, 0), NYg - 1);
  int flat = yi * NXg + xi;
  int slot = atomicAdd(&cnt[flat], 1);
  if (slot < MAXPTS) {
    float cx = (float)xi * 0.4f - 51.0f;  // X_MIN + X_RES/2
    float cy = (float)yi * 0.4f - 51.0f;
    float* d = dense + ((size_t)flat * MAXPTS + slot) * PDIM;
    d[0] = x; d[1] = y; d[2] = z; d[3] = it;
    d[4] = x - cx; d[5] = y - cy; d[6] = z;
  }
}

// ---------------------------------------------------------------------------
// Kernel 2: pack conv weights OIHW fp32 -> WMMA B-fragment order fp16:
//   dst[((tap*KS + ks)*(O/16) + ntile)*32 + lane][e]  with
//   n = ntile*16 + lane%16 ; k = ks*32 + (lane>>4)*16 + e
// ---------------------------------------------------------------------------
__global__ __launch_bounds__(256) void pack_weights_frag_kernel(
    const float* __restrict__ w, _Float16* __restrict__ dst, int O, int I) {
  int idx = blockIdx.x * 256 + threadIdx.x;
  int total = O * I * 9;
  if (idx >= total) return;
  int e = idx & 15;
  int lane = (idx >> 4) & 31;
  int ntile = (idx >> 9) % (O / 16);
  int rest = idx / (512 * (O / 16));
  int ks = rest % (I / 32);
  int tap = rest / (I / 32);
  int n = ntile * 16 + (lane & 15);
  int k = ks * 32 + (lane >> 4) * 16 + e;
  int ky = tap / 3, kx = tap % 3;
  dst[idx] = (_Float16)w[((size_t)(n * I + k) * 3 + ky) * 3 + kx];
}

// ---------------------------------------------------------------------------
// Kernel 3: PointNet. One wave per cell, 8 cells per block.
// ---------------------------------------------------------------------------
__global__ __launch_bounds__(256) void pointnet_kernel(
    const float* __restrict__ dense, const int* __restrict__ cnt,
    const float* __restrict__ w1, const float* __restrict__ s1,
    const float* __restrict__ t1, const float* __restrict__ w2,
    const float* __restrict__ s2, const float* __restrict__ t2,
    _Float16* __restrict__ bev) {  // NHWC: [cell][64]
  __shared__ float w1s[224], s1s[32], t1s[32], s2s[64], t2s[64];
  __shared__ __align__(16) _Float16 w2h[64][32];      // [n][k]
  __shared__ __align__(16) _Float16 h1ls[8][32][32];  // [wave][slot][ch]

  const int tid = threadIdx.x;
  if (tid < 224) w1s[tid] = w1[tid];
  if (tid < 32) { s1s[tid] = s1[tid]; t1s[tid] = t1[tid]; }
  if (tid < 64) { s2s[tid] = s2[tid]; t2s[tid] = t2[tid]; }
#pragma unroll
  for (int idx = tid; idx < 2048; idx += 256) {
    int nn = idx >> 5, kk = idx & 31;
    w2h[nn][kk] = (_Float16)w2[kk * 64 + nn];  // w2 is [32][64] K-major
  }
  __syncthreads();

  const int wv = tid >> 5;
  const int lane = tid & 31;
  const int cell = (blockIdx.x << 3) + wv;
  const int slot = lane;

  // layer 1 : 7 -> 32, per lane (lane = slot)
  const float* f = dense + ((size_t)cell * MAXPTS + slot) * PDIM;
  float fv[7];
#pragma unroll
  for (int k = 0; k < 7; ++k) fv[k] = f[k];
  float h[32];
#pragma unroll
  for (int j = 0; j < 32; ++j) {
    float acc = 0.f;
#pragma unroll
    for (int k = 0; k < 7; ++k) acc += fv[k] * w1s[k * 32 + j];
    h[j] = fmaxf(acc * s1s[j] + t1s[j], 0.f);
  }
#pragma unroll
  for (int q = 0; q < 4; ++q) {
    h8 v;
#pragma unroll
    for (int e = 0; e < 8; ++e) v[e] = (_Float16)h[q * 8 + e];
    *(h8*)&h1ls[wv][slot][q * 8] = v;
  }
  __syncthreads();

  // layer 2 : 32 -> 64 via WMMA. A = 32 slots x K32 (two M-tiles).
  const int ml = lane & 15, hi = lane >> 4;
  h16 a0 = ldA(&h1ls[wv][ml][0], hi);
  h16 a1 = ldA(&h1ls[wv][16 + ml][0], hi);

  v8f c0[4], c1[4];
#pragma unroll
  for (int t = 0; t < 4; ++t) {
    h16 b = ldB(&w2h[t * 16 + ml][0], hi);
    v8f z = {};
    c0[t] = __builtin_amdgcn_wmma_f32_16x16x32_f16(false, a0, false, b,
                                                   (short)0, z, false, false);
    c1[t] = __builtin_amdgcn_wmma_f32_16x16x32_f16(false, a1, false, b,
                                                   (short)0, z, false, false);
  }

  const int occ = cnt[cell];
#pragma unroll
  for (int t = 0; t < 4; ++t) {
    int ch = t * 16 + ml;
    float sc = s2s[ch], bi = t2s[ch];
    float m = 0.f;  // relu outputs are >= 0
#pragma unroll
    for (int r = 0; r < 8; ++r) {
      float u0 = fmaxf(c0[t][r] * sc + bi, 0.f);
      float u1 = fmaxf(c1[t][r] * sc + bi, 0.f);
      m = fmaxf(m, fmaxf(u0, u1));
    }
    float o = __shfl_xor(m, 16, 32);  // combine rows M and M+8
    m = fmaxf(m, o);
    if (lane < 16) {
      bev[(size_t)cell * 64 + ch] = (occ > 0) ? (_Float16)m : (_Float16)0.f;
    }
  }
}

// ---------------------------------------------------------------------------
// Kernel 4: implicit-GEMM 3x3 conv + BN + ReLU (NHWC fp16 in, WMMA).
//   Block tile: 64 pixels (x-run) x 64 out-channels, 8 waves = 4(M) x 2(N).
//   Whole 3x66xCIN patch staged in LDS once; no barriers in K loop.
// ---------------------------------------------------------------------------
template <int CIN, int COUT, bool OUTF32>
__global__ __launch_bounds__(256) void conv3x3_nhwc_kernel(
    const _Float16* __restrict__ in,   // NHWC [256][256][CIN]
    const _Float16* __restrict__ wpk,  // fragment-packed weights
    const float* __restrict__ scale, const float* __restrict__ bias,
    _Float16* __restrict__ out16,      // NHWC [256][256][COUT]
    float* __restrict__ out32) {       // NCHW [COUT][256][256]
  constexpr int CP = CIN + 8;          // x-stride pad -> conflict-free A reads
  constexpr int KSTEPS = CIN / 32;
  __shared__ __align__(16) _Float16 patch[3 * 66 * CP];

  const int tid = threadIdx.x;
  const int lane = tid & 31;
  const int wv = tid >> 5;
  const int wm = wv & 3;            // M sub-tile (0..3)
  const int wn = wv >> 2;           // N sub-tile (0..1)
  const int ml = lane & 15, hi = lane >> 4;

  const int mb = blockIdx.x;        // 1024 pixel-blocks of 64
  const int yo = mb >> 2;           // output row
  const int x0 = (mb & 3) << 6;     // output x base
  const int co0 = blockIdx.y << 6;  // out-channel base
  const int mbase = wm << 4;

  // ---- stage 3x66xCIN zero-padded input patch, coalesced b128 ----
  constexpr int CPERX = CIN / 8;
  constexpr int CHUNKS = 3 * 66 * CPERX;
  for (int c8 = tid; c8 < CHUNKS; c8 += 256) {
    int cc = (c8 % CPERX) * 8;
    int xx = (c8 / CPERX) % 66;
    int row = c8 / (CPERX * 66);
    int gx = x0 - 1 + xx;
    int gy = yo - 1 + row;
    h8 v = {};
    if (gx >= 0 && gx < 256 && gy >= 0 && gy < 256)
      v = *(const h8*)(in + (size_t)(gy * 256 + gx) * CIN + cc);
    *(h8*)&patch[(row * 66 + xx) * CP + cc] = v;
  }
  __syncthreads();

  v8f acc0 = {};
  v8f acc1 = {};
  const int ntg0 = (co0 >> 4) + (wn << 1);  // first of two 16-col B tiles

#pragma unroll
  for (int tap = 0; tap < 9; ++tap) {
    const int ky = tap / 3, kx = tap % 3;
    const _Float16* arow = &patch[(ky * 66 + mbase + ml + kx) * CP];
#pragma unroll
    for (int ks = 0; ks < KSTEPS; ++ks) {
      h16 a = ldA(arow + ks * 32, hi);
      const _Float16* bp =
          wpk +
          ((((size_t)tap * KSTEPS + ks) * (COUT / 16) + ntg0) * 32 + lane) * 16;
      __builtin_prefetch(bp + (size_t)(COUT / 16) * 512, 0, 1);
      h16 b0 = frag_combine(*(const h8*)bp, *(const h8*)(bp + 8));
      h16 b1 = frag_combine(*(const h8*)(bp + 512), *(const h8*)(bp + 520));
      acc0 = __builtin_amdgcn_wmma_f32_16x16x32_f16(false, a, false, b0,
                                                    (short)0, acc0, false,
                                                    false);
      acc1 = __builtin_amdgcn_wmma_f32_16x16x32_f16(false, a, false, b1,
                                                    (short)0, acc1, false,
                                                    false);
    }
  }

  // epilogue: BN + ReLU. C/D layout: row M = r + 8*hi, col N = lane%16
  const int pixrow = (mb << 6) + mbase + (hi << 3);
#pragma unroll
  for (int nt = 0; nt < 2; ++nt) {
    int ch = co0 + (wn << 5) + (nt << 4) + ml;
    float sc = scale[ch], bi = bias[ch];
    v8f a = nt ? acc1 : acc0;
#pragma unroll
    for (int r = 0; r < 8; ++r) {
      float v = fmaxf(a[r] * sc + bi, 0.f);
      int p = pixrow + r;
      if (OUTF32)
        out32[((size_t)ch << 16) + p] = v;          // final: NCHW fp32
      else
        out16[(size_t)p * COUT + ch] = (_Float16)v; // interior: NHWC fp16
    }
  }
}

// ---------------------------------------------------------------------------
// Launcher
// ---------------------------------------------------------------------------
extern "C" void kernel_launch(void* const* d_in, const int* in_sizes, int n_in,
                              void* d_out, int out_size, void* d_ws,
                              size_t ws_size, hipStream_t stream) {
  const float* points = (const float*)d_in[0];
  const float* w1 = (const float*)d_in[1];
  const float* s1 = (const float*)d_in[2];
  const float* t1 = (const float*)d_in[3];
  const float* w2 = (const float*)d_in[4];
  const float* s2 = (const float*)d_in[5];
  const float* t2 = (const float*)d_in[6];
  const float* cw1a = (const float*)d_in[7];
  const float* cs1a = (const float*)d_in[8];
  const float* ct1a = (const float*)d_in[9];
  const float* cw1b = (const float*)d_in[10];
  const float* cs1b = (const float*)d_in[11];
  const float* ct1b = (const float*)d_in[12];
  const float* cw2a = (const float*)d_in[13];
  const float* cs2a = (const float*)d_in[14];
  const float* ct2a = (const float*)d_in[15];
  const float* cw2b = (const float*)d_in[16];
  const float* cs2b = (const float*)d_in[17];
  const float* ct2b = (const float*)d_in[18];
  const int N = in_sizes[0] / 4;

  char* ws = (char*)d_ws;
  size_t off = 0;
  auto alloc = [&](size_t bytes) -> char* {
    char* p = ws + off;
    off = (off + bytes + 255) & ~(size_t)255;
    return p;
  };
  int* cnt = (int*)alloc((size_t)NCELLSg * 4);
  float* dense = (float*)alloc((size_t)NCELLSg * MAXPTS * PDIM * 4);
  _Float16* bev = (_Float16*)alloc((size_t)64 * NCELLSg * 2);
  _Float16* c1b = (_Float16*)alloc((size_t)128 * NCELLSg * 2);
  _Float16* c2b = (_Float16*)alloc((size_t)128 * NCELLSg * 2);
  _Float16* c3b = (_Float16*)alloc((size_t)64 * NCELLSg * 2);
  _Float16* wa = (_Float16*)alloc((size_t)9 * 128 * 64 * 2);
  _Float16* wb = (_Float16*)alloc((size_t)9 * 128 * 128 * 2);
  _Float16* wc = (_Float16*)alloc((size_t)9 * 64 * 128 * 2);
  _Float16* wd = (_Float16*)alloc((size_t)9 * 64 * 64 * 2);

  hipMemsetAsync(cnt, 0, (size_t)NCELLSg * 4, stream);
  hipMemsetAsync(dense, 0, (size_t)NCELLSg * MAXPTS * PDIM * 4, stream);

  pack_weights_frag_kernel<<<(9 * 128 * 64 + 255) / 256, 256, 0, stream>>>(
      cw1a, wa, 128, 64);
  pack_weights_frag_kernel<<<(9 * 128 * 128 + 255) / 256, 256, 0, stream>>>(
      cw1b, wb, 128, 128);
  pack_weights_frag_kernel<<<(9 * 64 * 128 + 255) / 256, 256, 0, stream>>>(
      cw2a, wc, 64, 128);
  pack_weights_frag_kernel<<<(9 * 64 * 64 + 255) / 256, 256, 0, stream>>>(
      cw2b, wd, 64, 64);

  pillarize_kernel<<<(N + 255) / 256, 256, 0, stream>>>(points, N, cnt, dense);

  pointnet_kernel<<<NCELLSg / 8, 256, 0, stream>>>(dense, cnt, w1, s1, t1, w2,
                                                   s2, t2, bev);

  conv3x3_nhwc_kernel<64, 128, false>
      <<<dim3(1024, 2), 256, 0, stream>>>(bev, wa, cs1a, ct1a, c1b, nullptr);
  conv3x3_nhwc_kernel<128, 128, false>
      <<<dim3(1024, 2), 256, 0, stream>>>(c1b, wb, cs1b, ct1b, c2b, nullptr);
  conv3x3_nhwc_kernel<128, 64, false>
      <<<dim3(1024, 1), 256, 0, stream>>>(c2b, wc, cs2a, ct2a, c3b, nullptr);
  conv3x3_nhwc_kernel<64, 64, true><<<dim3(1024, 1), 256, 0, stream>>>(
      c3b, wd, cs2b, ct2b, nullptr, (float*)d_out);
}
